// MultiHeadAttention_3281355014158
// MI455X (gfx1250) — compile-verified, matrix operand order checked by reading, not run on hardware
//
#include <hip/hip_runtime.h>

#define D_MODEL 1024
#define NHEAD   16
#define DK      64
#define SEQ     2048
#define BATCH   4
#define ROWS    (BATCH * SEQ)      // 8192
#define NEGC    (-1.0e9f)
#define LOG2E   1.44269504088896340736f
#define KT      64                 // key tile in attention

typedef __attribute__((ext_vector_type(16))) __bf16 bf16x16;
typedef __attribute__((ext_vector_type(8)))  float  f32x8;

union FragU { bf16x16 v; uint4 q[2]; };

__device__ __forceinline__ f32x8 wmma_bf16(bf16x16 a, bf16x16 b, f32x8 c) {
  // D(16x16 f32) = A(16x32 bf16) * B(32x16 bf16) + C
  return __builtin_amdgcn_wmma_f32_16x16x32_bf16(false, a, false, b, (short)0, c,
                                                 false, false);
}

// Single-instruction lane exchange: DS_SWIZZLE_B32 group-of-32 XOR mode.
// offset[15]=0, and_mask=0x1f, or_mask=0, xor_mask=MASK.
template <int MASK>
__device__ __forceinline__ float swz_xor(float x) {
  int r = __builtin_amdgcn_ds_swizzle(__float_as_int(x), (MASK << 10) | 0x1f);
  return __int_as_float(r);
}

// 16-bit A-layout fragment load from a row-major LDS tile.
// lane 0-15 : row = row0+lane,    K = {k0..k0+7, k0+16..k0+23}
// lane 16-31: row = row0+lane-16, K = {k0+8..k0+15, k0+24..k0+31}
// Also used for B fragments whenever the B-matrix N dimension is the LDS row.
__device__ __forceinline__ bf16x16 load_frag_row(const __bf16* lds, int stride,
                                                 int row0, int k0, int lane) {
  int r  = row0 + (lane & 15);
  int kb = k0 + ((lane >> 4) << 3);
  FragU f;
  f.q[0] = *(const uint4*)(lds + r * stride + kb);
  f.q[1] = *(const uint4*)(lds + r * stride + kb + 16);
  return f.v;
}

// CDNA5 async global->LDS copy, 16 bytes per lane (GVS addressing mode).
// ldsoff: byte offset of destination inside the workgroup's LDS allocation
// (low 32 bits of a generic __shared__ pointer per the flat aperture rule).
__device__ __forceinline__ void async_ld16(unsigned ldsoff, const void* base,
                                           unsigned goff) {
  asm volatile("global_load_async_to_lds_b128 %0, %1, %2 offset:0"
               :: "v"(ldsoff), "v"(goff), "s"(base) : "memory");
}
__device__ __forceinline__ void wait_async0() {
  asm volatile("s_wait_asynccnt 0x0" ::: "memory");
}

// ---------------------------------------------------------------------------
// Kernel 0: RoPE cos/sin table, one entry per (pos, freq) pair.
// 65536 sincos evaluations instead of ~16M in the GEMM epilogue.
// ---------------------------------------------------------------------------
__global__ __launch_bounds__(256)
void rope_table_kernel(float* __restrict__ ct, float* __restrict__ st)
{
  int t = blockIdx.x * 256 + threadIdx.x;      // 0 .. SEQ*32-1
  int pos = t >> 5, i = t & 31;
  float invf = __powf(10000.0f, -(float)(2 * i) * (1.0f / DK));
  float th = (float)pos * invf;
  float sn, cs;
  __sincosf(th, &sn, &cs);
  ct[t] = cs;
  st[t] = sn;
}

// ---------------------------------------------------------------------------
// Kernel 1: fused Q/K/V projection GEMM (x @ W^T + b) with RoPE epilogue.
// grid = (N_tiles=8, M_tiles=64, z=3 for q/k/v), block = 256.
// Q,K stored [b][h][s][d]; V stored TRANSPOSED [b][h][d][s] so the attention
// kernel's P*V B-fragment becomes a contiguous row load.
// ---------------------------------------------------------------------------
__global__ __launch_bounds__(256)
void qkv_gemm_rope_kernel(const float* __restrict__ x,
                          const float* __restrict__ wq, const float* __restrict__ bq,
                          const float* __restrict__ wk, const float* __restrict__ bk,
                          const float* __restrict__ wv, const float* __restrict__ bv,
                          const float* __restrict__ ct, const float* __restrict__ st,
                          __bf16* __restrict__ Qo, __bf16* __restrict__ Ko,
                          __bf16* __restrict__ Vo)
{
  __shared__ __bf16 As[128 * 40];   // 128 rows x 32 k, padded to 40 (80B, 16B-aligned)
  __shared__ __bf16 Bs[128 * 40];

  const int tid  = threadIdx.x;
  const int lane = tid & 31;
  const int w    = tid >> 5;
  const int wm   = w >> 1;          // 0..3  -> 32-row strip
  const int wn   = w & 1;           // 0..1  -> 64-col strip
  const int Mb   = blockIdx.y * 128;
  const int Nb   = blockIdx.x * 128;
  const int z    = blockIdx.z;

  const float* wsel = (z == 0) ? wq : (z == 1) ? wk : wv;
  const float* bsel = (z == 0) ? bq : (z == 1) ? bk : bv;
  __bf16*      osel = (z == 0) ? Qo : (z == 1) ? Ko : Vo;

  f32x8 acc[2][4] = {};

  const int lr = tid >> 1;          // 0..127
  const int lc = (tid & 1) * 16;    // 0 or 16

  for (int kc = 0; kc < D_MODEL; kc += 32) {
    __syncthreads();
    {
      const float4* ax = (const float4*)(x    + (size_t)(Mb + lr) * D_MODEL + kc + lc);
      const float4* bx = (const float4*)(wsel + (size_t)(Nb + lr) * D_MODEL + kc + lc);
#pragma unroll
      for (int i = 0; i < 4; ++i) {
        float4 fa = ax[i], fb = bx[i];
        __bf16* da = As + lr * 40 + lc + i * 4;
        __bf16* db = Bs + lr * 40 + lc + i * 4;
        da[0] = (__bf16)fa.x; da[1] = (__bf16)fa.y; da[2] = (__bf16)fa.z; da[3] = (__bf16)fa.w;
        db[0] = (__bf16)fb.x; db[1] = (__bf16)fb.y; db[2] = (__bf16)fb.z; db[3] = (__bf16)fb.w;
      }
    }
    __syncthreads();

    bf16x16 a0 = load_frag_row(As, 40, wm * 32 + 0,  0, lane);
    bf16x16 a1 = load_frag_row(As, 40, wm * 32 + 16, 0, lane);
#pragma unroll
    for (int n_ = 0; n_ < 4; ++n_) {
      bf16x16 bfr = load_frag_row(Bs, 40, wn * 64 + n_ * 16, 0, lane);
      acc[0][n_] = wmma_bf16(a0, bfr, acc[0][n_]);
      acc[1][n_] = wmma_bf16(a1, bfr, acc[1][n_]);
    }
  }

  const float sc       = (z == 0) ? 0.125f : 1.0f;  // fold 1/sqrt(dk) into Q
  const int   halfofs  = (lane >> 4) << 3;
  const bool  evenlane = (lane & 1) == 0;           // n parity == lane parity

  float biasn[4];
  int   itab[4];
#pragma unroll
  for (int n_ = 0; n_ < 4; ++n_) {
    int n = Nb + wn * 64 + n_ * 16 + (lane & 15);
    biasn[n_] = bsel[n];
    itab[n_]  = (n & (DK - 1)) >> 1;               // RoPE frequency index
  }

#pragma unroll
  for (int im = 0; im < 2; ++im) {
#pragma unroll
    for (int n_ = 0; n_ < 4; ++n_) {
#pragma unroll
      for (int jr = 0; jr < 8; ++jr) {
        int m = Mb + wm * 32 + im * 16 + jr + halfofs;
        int n = Nb + wn * 64 + n_ * 16 + (lane & 15);
        float val = acc[im][n_][jr] + biasn[n_];
        if (z < 2) {
          // RoPE: the (even,odd) head-dim pair sits in adjacent lanes.
          int pos = m & (SEQ - 1);
          int ti  = (pos << 5) + itab[n_];
          float cs = ct[ti];
          float sn = st[ti];
          float other = swz_xor<1>(val);
          val = evenlane ? (cs * val - sn * other)
                         : (sn * other + cs * val);
        }
        val *= sc;
        int b_ = m >> 11, s_ = m & (SEQ - 1), h_ = n >> 6, d_ = n & 63;
        size_t idx;
        if (z == 2)  // V transposed: [b][h][d][s]
          idx = ((size_t)((b_ * NHEAD + h_) * DK + d_) << 11) + s_;
        else         // Q,K: [b][h][s][d]
          idx = ((size_t)((b_ * NHEAD + h_) * SEQ + s_) << 6) + d_;
        osel[idx] = (__bf16)val;
      }
    }
  }
}

// ---------------------------------------------------------------------------
// Kernel 2: flash-attention over 128-query tiles, streaming 64-key tiles.
// grid = (16 q-tiles, 64 b*h), block = 256. One wave owns 16 query rows.
// All global->LDS staging uses GLOBAL_LOAD_ASYNC_TO_LDS_B128 (ASYNCcnt).
// ---------------------------------------------------------------------------
__global__ __launch_bounds__(256)
void attn_kernel(const __bf16* __restrict__ Q, const __bf16* __restrict__ K,
                 const __bf16* __restrict__ Vt, __bf16* __restrict__ AO)
{
  __shared__ __bf16 Qs[128 * 72];   // 18 KB   [qrow][d]
  __shared__ __bf16 Ks[KT * 72];    //  9 KB   [krow][d]
  __shared__ __bf16 Vs[DK * 72];    //  9 KB   [d][krow]  (V transposed)
  __shared__ __bf16 Ps[128 * 72];   // 18 KB   [qrow][krow]

  const int tid = threadIdx.x, lane = tid & 31, w = tid >> 5;
  const int qt = blockIdx.x, bh = blockIdx.y;
  const int q0 = qt * 128;
  const int m0 = w * 16;
  const int halfofs = (lane >> 4) << 3;

  const __bf16* Qb  = Q  + (size_t)bh * SEQ * DK;
  const __bf16* Kb  = K  + (size_t)bh * SEQ * DK;
  const __bf16* Vtb = Vt + (size_t)bh * DK * SEQ;

  { // async-load the 128x64 query tile once
    int r = tid >> 1, cb = (tid & 1) * 32;
    unsigned g0 = (unsigned)(((q0 + r) * DK + cb) * 2);
    unsigned l0 = (unsigned)(uintptr_t)(Qs + r * 72 + cb);
#pragma unroll
    for (int i = 0; i < 4; ++i)
      async_ld16(l0 + i * 16, Qb, g0 + i * 16);
  }

  f32x8 o[4] = {};
  float mrow[8], lrow[8];
#pragma unroll
  for (int i = 0; i < 8; ++i) { mrow[i] = -3.0e38f; lrow[i] = 0.0f; }

  const int jmax = 2 * qt + 1;   // causal: only key tiles <= query tile
  for (int j = 0; j <= jmax; ++j) {
    __syncthreads();             // previous iteration done reading Ks/Vs/Ps
    { // async-load 64x64 K tile and transposed 64x64 V tile
      int r = tid >> 2, cb = (tid & 3) * 16;
      unsigned gk = (unsigned)(((j * KT + r) * DK + cb) * 2);
      unsigned lk = (unsigned)(uintptr_t)(Ks + r * 72 + cb);
      async_ld16(lk,      Kb, gk);
      async_ld16(lk + 16, Kb, gk + 16);
      unsigned gv = (unsigned)((r * SEQ + j * KT + cb) * 2);  // r = d row
      unsigned lv = (unsigned)(uintptr_t)(Vs + r * 72 + cb);
      async_ld16(lv,      Vtb, gv);
      async_ld16(lv + 16, Vtb, gv + 16);
    }
    wait_async0();
    __syncthreads();

    // S = Q * K^T  (16 rows x 64 keys per wave), Q pre-scaled by 1/sqrt(dk)
    f32x8 sfr[4] = {};
#pragma unroll
    for (int kk = 0; kk < 2; ++kk) {
      bf16x16 a = load_frag_row(Qs, 72, m0, kk * 32, lane);
#pragma unroll
      for (int n_ = 0; n_ < 4; ++n_) {
        bf16x16 b = load_frag_row(Ks, 72, n_ * 16, kk * 32, lane);
        sfr[n_] = wmma_bf16(a, b, sfr[n_]);
      }
    }
    // causal mask (additive NEG, matching the reference)
#pragma unroll
    for (int n_ = 0; n_ < 4; ++n_) {
      int col = j * KT + n_ * 16 + (lane & 15);
#pragma unroll
      for (int jr = 0; jr < 8; ++jr) {
        int row = q0 + m0 + jr + halfofs;
        if (col > row) sfr[n_][jr] += NEGC;
      }
    }
    // online softmax; XOR {1,2,4,8} swizzles stay inside the 16-lane N-group
#pragma unroll
    for (int jr = 0; jr < 8; ++jr) {
      float nm = mrow[jr];
#pragma unroll
      for (int n_ = 0; n_ < 4; ++n_) nm = fmaxf(nm, sfr[n_][jr]);
      nm = fmaxf(nm, swz_xor<1>(nm));
      nm = fmaxf(nm, swz_xor<2>(nm));
      nm = fmaxf(nm, swz_xor<4>(nm));
      nm = fmaxf(nm, swz_xor<8>(nm));
      float alpha = exp2f((mrow[jr] - nm) * LOG2E);
      float ss = 0.0f;
#pragma unroll
      for (int n_ = 0; n_ < 4; ++n_) {
        float p = exp2f((sfr[n_][jr] - nm) * LOG2E);
        sfr[n_][jr] = p;
        ss += p;
      }
      ss += swz_xor<1>(ss);
      ss += swz_xor<2>(ss);
      ss += swz_xor<4>(ss);
      ss += swz_xor<8>(ss);
      lrow[jr] = lrow[jr] * alpha + ss;
      mrow[jr] = nm;
#pragma unroll
      for (int n_ = 0; n_ < 4; ++n_) o[n_][jr] *= alpha;
    }
    // stage P through LDS to re-enter A-layout
#pragma unroll
    for (int n_ = 0; n_ < 4; ++n_)
#pragma unroll
      for (int jr = 0; jr < 8; ++jr)
        Ps[(m0 + jr + halfofs) * 72 + n_ * 16 + (lane & 15)] = (__bf16)sfr[n_][jr];
    __syncthreads();

    // O += P * V   (B-fragment: Vs rows are d (=N), columns are key (=K))
    bf16x16 pa0 = load_frag_row(Ps, 72, m0, 0,  lane);
    bf16x16 pa1 = load_frag_row(Ps, 72, m0, 32, lane);
#pragma unroll
    for (int n_ = 0; n_ < 4; ++n_) {
      bf16x16 b0 = load_frag_row(Vs, 72, n_ * 16, 0,  lane);
      o[n_] = wmma_bf16(pa0, b0, o[n_]);
      bf16x16 b1 = load_frag_row(Vs, 72, n_ * 16, 32, lane);
      o[n_] = wmma_bf16(pa1, b1, o[n_]);
    }
  }

  // normalize and write attention output as [b][s][h*64+d] bf16
  const int b_ = bh >> 4, h_ = bh & 15;
#pragma unroll
  for (int jr = 0; jr < 8; ++jr) {
    float inv = 1.0f / lrow[jr];
    int row = q0 + m0 + jr + halfofs;
#pragma unroll
    for (int n_ = 0; n_ < 4; ++n_) {
      int dcol = n_ * 16 + (lane & 15);
      AO[(size_t)(b_ * SEQ + row) * D_MODEL + h_ * DK + dcol] =
          (__bf16)(o[n_][jr] * inv);
    }
  }
}

// ---------------------------------------------------------------------------
// Kernel 3: output projection (AO @ wo^T + bo) -> f32.
// grid = (8, 64), block = 256. A tile staged with async LDS loads.
// ---------------------------------------------------------------------------
__global__ __launch_bounds__(256)
void out_proj_kernel(const __bf16* __restrict__ AO, const float* __restrict__ wo,
                     const float* __restrict__ bo, float* __restrict__ out)
{
  __shared__ __bf16 As[128 * 40];
  __shared__ __bf16 Bs[128 * 40];

  const int tid = threadIdx.x, lane = tid & 31, w = tid >> 5;
  const int wm = w >> 1, wn = w & 1;
  const int Mb = blockIdx.y * 128, Nb = blockIdx.x * 128;

  f32x8 acc[2][4] = {};

  const int lr = tid >> 1, lc = (tid & 1) * 16;
  for (int kc = 0; kc < D_MODEL; kc += 32) {
    __syncthreads();
    {
      // A tile is already bf16: pure byte copy -> async global->LDS
      unsigned ga = (unsigned)((((Mb + lr) * D_MODEL) + kc + lc) * 2);
      unsigned la = (unsigned)(uintptr_t)(As + lr * 40 + lc);
      async_ld16(la,      AO, ga);
      async_ld16(la + 16, AO, ga + 16);
      // B tile needs f32->bf16 conversion: manual staging
      const float4* bx = (const float4*)(wo + (size_t)(Nb + lr) * D_MODEL + kc + lc);
#pragma unroll
      for (int i = 0; i < 4; ++i) {
        float4 fb = bx[i];
        __bf16* db = Bs + lr * 40 + lc + i * 4;
        db[0] = (__bf16)fb.x; db[1] = (__bf16)fb.y; db[2] = (__bf16)fb.z; db[3] = (__bf16)fb.w;
      }
    }
    wait_async0();
    __syncthreads();

    bf16x16 a0 = load_frag_row(As, 40, wm * 32 + 0,  0, lane);
    bf16x16 a1 = load_frag_row(As, 40, wm * 32 + 16, 0, lane);
#pragma unroll
    for (int n_ = 0; n_ < 4; ++n_) {
      bf16x16 bfr = load_frag_row(Bs, 40, wn * 64 + n_ * 16, 0, lane);
      acc[0][n_] = wmma_bf16(a0, bfr, acc[0][n_]);
      acc[1][n_] = wmma_bf16(a1, bfr, acc[1][n_]);
    }
  }

  const int halfofs = (lane >> 4) << 3;
  float biasn[4];
#pragma unroll
  for (int n_ = 0; n_ < 4; ++n_)
    biasn[n_] = bo[Nb + wn * 64 + n_ * 16 + (lane & 15)];

#pragma unroll
  for (int im = 0; im < 2; ++im) {
#pragma unroll
    for (int n_ = 0; n_ < 4; ++n_) {
#pragma unroll
      for (int jr = 0; jr < 8; ++jr) {
        int m = Mb + wm * 32 + im * 16 + jr + halfofs;
        int n = Nb + wn * 64 + n_ * 16 + (lane & 15);
        out[(size_t)m * D_MODEL + n] = acc[im][n_][jr] + biasn[n_];
      }
    }
  }
}

// ---------------------------------------------------------------------------
extern "C" void kernel_launch(void* const* d_in, const int* in_sizes, int n_in,
                              void* d_out, int out_size, void* d_ws, size_t ws_size,
                              hipStream_t stream)
{
  (void)in_sizes; (void)n_in; (void)out_size; (void)ws_size;
  const float* x  = (const float*)d_in[0];
  const float* wq = (const float*)d_in[1];
  const float* bq = (const float*)d_in[2];
  const float* wk = (const float*)d_in[3];
  const float* bk = (const float*)d_in[4];
  const float* wv = (const float*)d_in[5];
  const float* bv = (const float*)d_in[6];
  const float* wo = (const float*)d_in[7];
  const float* bo = (const float*)d_in[8];

  const size_t elems = (size_t)ROWS * D_MODEL;   // 8M elements per buffer
  __bf16* Q  = (__bf16*)d_ws;
  __bf16* K  = Q + elems;
  __bf16* Vt = K + elems;
  __bf16* AO = Vt + elems;                       // 64 MB of bf16 buffers
  float*  ct = (float*)(AO + elems);             // + 2 x 256 KB RoPE tables
  float*  st = ct + (size_t)SEQ * 32;

  rope_table_kernel<<<dim3(SEQ * 32 / 256), 256, 0, stream>>>(ct, st);
  qkv_gemm_rope_kernel<<<dim3(8, 64, 3), 256, 0, stream>>>(x, wq, bq, wk, bk,
                                                           wv, bv, ct, st,
                                                           Q, K, Vt);
  attn_kernel<<<dim3(16, 64), 256, 0, stream>>>(Q, K, Vt, AO);
  out_proj_kernel<<<dim3(8, 64), 256, 0, stream>>>(AO, wo, bo, (float*)d_out);
}
